// RNNSearchAttention_63539746177420
// MI455X (gfx1250) — matrix operand order repaired and checked by reading, not compiled, not run on hardware
//
#include <hip/hip_runtime.h>
#include <hip/hip_bf16.h>
#include <math.h>

// ---------------------------------------------------------------------------
// RNNSearch (Bahdanau) attention, fused for MI455X (gfx1250, wave32, WMMA).
// S=4096, B=16, E=1024, D=1024, A=512.
// Dominant GEMM (enc@We, 68.7 GFLOP) runs on v_wmma_f32_16x16x32_bf16 with the
// tanh/dot-Ws epilogue fused. Each block processes TWO timesteps so every
// Wt (B-matrix) fragment load feeds two WMMAs -> halves L2 traffic.
// ---------------------------------------------------------------------------

typedef __attribute__((ext_vector_type(16))) __bf16 v16bf;
typedef __attribute__((ext_vector_type(8)))  __bf16 v8bf;
typedef __attribute__((ext_vector_type(8)))  float  v8f;

#define S_DIM 4096
#define B_DIM 16
#define E_DIM 1024
#define A_DIM 512
#define LDS_STRIDE (E_DIM + 8)          // +16B row pad -> conflict-free banks
#define STRIPE     (B_DIM * LDS_STRIDE) // one timestep's enc stripe (bf16)
#define LDS_BYTES  (2 * STRIPE * 2)     // two stripes, dynamic LDS (66 KB)

// Branch-free tanh: avoids the exec-mask branchy libm expansion.
__device__ __forceinline__ float tanh_fast(float x) {
  float ax = fminf(fabsf(x), 15.0f);
  float e  = __expf(2.0f * ax);               // v_exp_f32 based
  float t  = __fdividef(e - 1.0f, e + 1.0f);
  return copysignf(t, x);
}

// ---------------------------------------------------------------------------
// Kernel 0: We [E][A] f32  ->  Wt [A][E] bf16 (transposed, K-contiguous rows)
// ---------------------------------------------------------------------------
__global__ __launch_bounds__(256) void prep_wt(const float* __restrict__ We,
                                               __bf16* __restrict__ Wt) {
  int i = blockIdx.x * 256 + threadIdx.x;      // over A_DIM*E_DIM
  int n = i >> 10;                             // row of Wt (column of We)
  int e = i & (E_DIM - 1);
  Wt[i] = (__bf16)We[(size_t)e * A_DIM + n];
}

// ---------------------------------------------------------------------------
// Kernel 1: wdb[b][n] = dec[b]·Wd[:,n] + bd[n] + be[n]   (16x512, tiny)
// ---------------------------------------------------------------------------
__global__ __launch_bounds__(256) void prep_wdb(const float* __restrict__ dec,
                                                const float* __restrict__ Wd,
                                                const float* __restrict__ bd,
                                                const float* __restrict__ be,
                                                float* __restrict__ wdb) {
  int i = blockIdx.x * 256 + threadIdx.x;      // over B_DIM*A_DIM
  int b = i >> 9;
  int n = i & (A_DIM - 1);
  float acc = bd[n] + be[n];
  const float* dp = dec + (size_t)b * 1024;    // D = 1024
  #pragma unroll 4
  for (int k = 0; k < 1024; ++k)
    acc += dp[k] * Wd[(size_t)k * A_DIM + n];
  wdb[i] = acc;
}

// ---------------------------------------------------------------------------
// Kernel 2: fused scores. 1 block per PAIR of timesteps (128 thr = 4 waves).
//   scores[s][b] = sum_n tanh( (enc[s]@We)[b][n] + wdb[b][n] ) * Ws[n]
// Each wave owns a 128-column slice of A for BOTH timesteps: every B-fragment
// load is consumed by two v_wmma, halving Wt/L2 traffic.
// ---------------------------------------------------------------------------
__global__ __launch_bounds__(128) void score_kernel(const float* __restrict__ enc,
                                                    const __bf16* __restrict__ Wt,
                                                    const float* __restrict__ wdb,
                                                    const float* __restrict__ Ws,
                                                    float* __restrict__ scores) {
  extern __shared__ __align__(16) char smem_raw[];
  __bf16* lds_a = (__bf16*)smem_raw;           // [2][B_DIM][LDS_STRIDE]
  __shared__ float lds_red[2][4][16];

  const int s0   = blockIdx.x * 2;             // first of the two timesteps
  const int tid  = threadIdx.x;
  const int wave = tid >> 5;
  const int lane = tid & 31;
  const int halfSel = lane >> 4;               // 0: lanes 0-15, 1: lanes 16-31
  const int row     = lane & 15;               // A row (=batch) / B,C column

  // ---- stage 2 enc stripes (2 x 16 x 1024 f32 -> bf16), coalesced ----
  const float4* encs =
      reinterpret_cast<const float4*>(enc + (size_t)s0 * B_DIM * E_DIM);
  for (int i = tid; i < 2 * B_DIM * E_DIM / 4; i += 128) {
    float4 v = encs[i];
    int e0 = (i & 4095) * 4;                   // within one stripe
    int sl = i >> 12;                          // which stripe
    int b  = e0 >> 10;
    int e  = e0 & (E_DIM - 1);
    __bf16* dst = &lds_a[sl * STRIPE + b * LDS_STRIDE + e];
    dst[0] = (__bf16)v.x; dst[1] = (__bf16)v.y;
    dst[2] = (__bf16)v.z; dst[3] = (__bf16)v.w;
  }
  __syncthreads();

  float sp[2][8];
  #pragma unroll
  for (int sl = 0; sl < 2; ++sl)
    #pragma unroll
    for (int r = 0; r < 8; ++r) sp[sl][r] = 0.f;

  const int nbase = wave * 128;                // this wave's 128 columns of A
  for (int g = 0; g < 2; ++g) {                // 2 groups of 4 N-tiles
    const int n0 = nbase + g * 64;
    v8f acc[2][4];
    #pragma unroll
    for (int sl = 0; sl < 2; ++sl)
      #pragma unroll
      for (int j = 0; j < 4; ++j) acc[sl][j] = (v8f){};

    for (int k0 = 0; k0 < E_DIM; k0 += 32) {
      // A fragments (16x32 bf16), one per timestep. Layout: lane<16 holds
      // K 0-15, lane>=16 holds K 16-31 (matches B); row M = lane%16.
      v16bf Af[2];
      #pragma unroll
      for (int sl = 0; sl < 2; ++sl) {
        const __bf16* arow =
            &lds_a[sl * STRIPE + row * LDS_STRIDE + k0 + halfSel * 16];
        v8bf alo = *reinterpret_cast<const v8bf*>(arow);
        v8bf ahi = *reinterpret_cast<const v8bf*>(arow + 8);
        #pragma unroll
        for (int i = 0; i < 8; ++i) { Af[sl][i] = alo[i]; Af[sl][i + 8] = ahi[i]; }
      }

      #pragma unroll
      for (int j = 0; j < 4; ++j) {
        // B fragment (32x16 bf16): col N = lane%16; lanes 0-15 K 0-15,
        // lanes 16-31 K 16-31; Wt rows are K-contiguous -> 2x16B loads.
        const __bf16* bp = Wt + (size_t)(n0 + j * 16 + row) * E_DIM
                              + k0 + halfSel * 16;
        v8bf blo = *reinterpret_cast<const v8bf*>(bp);
        v8bf bhi = *reinterpret_cast<const v8bf*>(bp + 8);
        v16bf Bf;
        #pragma unroll
        for (int i = 0; i < 8; ++i) { Bf[i] = blo[i]; Bf[i + 8] = bhi[i]; }

        acc[0][j] = __builtin_amdgcn_wmma_f32_16x16x32_bf16(
            false, Af[0], false, Bf, (short)0, acc[0][j], false, false);
        acc[1][j] = __builtin_amdgcn_wmma_f32_16x16x32_bf16(
            false, Af[1], false, Bf, (short)0, acc[1][j], false, false);
      }
    }

    // ---- fused epilogue: + wdb, tanh, * Ws, accumulate per-row score ----
    #pragma unroll
    for (int j = 0; j < 4; ++j) {
      const int ncol = n0 + j * 16 + row;      // C: col N = lane%16
      const float wsv = Ws[ncol];
      #pragma unroll
      for (int r = 0; r < 8; ++r) {            // C: row M = r + 8*halfSel
        const int b = r + halfSel * 8;
        const float w = wdb[b * A_DIM + ncol];
        sp[0][r] += tanh_fast(acc[0][j][r] + w) * wsv;
        sp[1][r] += tanh_fast(acc[1][j][r] + w) * wsv;
      }
    }
  }

  // reduce over the 16 columns (lanes within each 16-lane half)
  #pragma unroll
  for (int sl = 0; sl < 2; ++sl) {
    #pragma unroll
    for (int r = 0; r < 8; ++r) {
      float v = sp[sl][r];
      #pragma unroll
      for (int m = 1; m < 16; m <<= 1) v += __shfl_xor(v, m, 32);
      sp[sl][r] = v;
    }
    if (row == 0) {
      #pragma unroll
      for (int r = 0; r < 8; ++r) lds_red[sl][wave][halfSel * 8 + r] = sp[sl][r];
    }
  }
  __syncthreads();
  if (tid < 32) {                              // sum the 4 waves' column slices
    const int sl = tid >> 4, b = tid & 15;
    float tot = lds_red[sl][0][b] + lds_red[sl][1][b]
              + lds_red[sl][2][b] + lds_red[sl][3][b];
    scores[(s0 + sl) * B_DIM + b] = tot;       // (+bs dropped: softmax-invariant)
  }
}

// ---------------------------------------------------------------------------
// Kernel 3: softmax over S, per batch column, in place.
// ---------------------------------------------------------------------------
__global__ __launch_bounds__(256) void softmax_kernel(float* __restrict__ sc) {
  __shared__ float red[256];
  const int b = blockIdx.x, tid = threadIdx.x;

  float m = -INFINITY;
  for (int s = tid; s < S_DIM; s += 256) m = fmaxf(m, sc[s * B_DIM + b]);
  red[tid] = m; __syncthreads();
  for (int off = 128; off > 0; off >>= 1) {
    if (tid < off) red[tid] = fmaxf(red[tid], red[tid + off]);
    __syncthreads();
  }
  m = red[0]; __syncthreads();

  float sum = 0.f;
  for (int s = tid; s < S_DIM; s += 256) sum += __expf(sc[s * B_DIM + b] - m);
  red[tid] = sum; __syncthreads();
  for (int off = 128; off > 0; off >>= 1) {
    if (tid < off) red[tid] += red[tid + off];
    __syncthreads();
  }
  const float inv = 1.0f / red[0];

  for (int s = tid; s < S_DIM; s += 256)
    sc[s * B_DIM + b] = __expf(sc[s * B_DIM + b] - m) * inv;
}

// ---------------------------------------------------------------------------
// Kernel 4: partial contexts over S-chunks (deterministic, no FP atomics).
//   grid = (4 e-chunks, 16 b, 8 s-chunks), 256 thr.
// ---------------------------------------------------------------------------
__global__ __launch_bounds__(256) void ctx_partial(const float* __restrict__ enc,
                                                   const float* __restrict__ alpha,
                                                   float* __restrict__ part) {
  const int e  = blockIdx.x * 256 + threadIdx.x;
  const int b  = blockIdx.y;
  const int sc = blockIdx.z;
  float acc = 0.f;
  const int s0 = sc * (S_DIM / 8);
  #pragma unroll 4
  for (int s = s0; s < s0 + S_DIM / 8; ++s)
    acc += alpha[s * B_DIM + b] * enc[((size_t)s * B_DIM + b) * E_DIM + e];
  part[((size_t)sc * B_DIM + b) * E_DIM + e] = acc;
}

__global__ __launch_bounds__(256) void ctx_reduce(const float* __restrict__ part,
                                                  float* __restrict__ out) {
  const int i = blockIdx.x * 256 + threadIdx.x;   // B_DIM*E_DIM = 16384
  float a = 0.f;
  #pragma unroll
  for (int c = 0; c < 8; ++c) a += part[(size_t)c * B_DIM * E_DIM + i];
  out[i] = a;
}

// ---------------------------------------------------------------------------
extern "C" void kernel_launch(void* const* d_in, const int* in_sizes, int n_in,
                              void* d_out, int out_size, void* d_ws, size_t ws_size,
                              hipStream_t stream) {
  const float* enc = (const float*)d_in[0];   // [S,B,E]
  const float* dec = (const float*)d_in[1];   // [1,B,D]
  const float* We  = (const float*)d_in[2];   // [E,A]
  const float* be  = (const float*)d_in[3];   // [A]
  const float* Wd  = (const float*)d_in[4];   // [D,A]
  const float* bd  = (const float*)d_in[5];   // [A]
  const float* Ws  = (const float*)d_in[6];   // [A,1]
  // d_in[7] = bs: constant over S -> cancels in softmax, unused.
  float* out = (float*)d_out;                 // [B,E]

  char* ws = (char*)d_ws;
  __bf16* Wt     = (__bf16*)ws;                                   // 1 MB
  float*  wdb    = (float*)(ws + (1u << 20));                     // 32 KB
  float*  scores = (float*)(ws + (1u << 20) + (32u << 10));       // 256 KB
  float*  part   = (float*)(ws + (1u << 20) + (288u << 10));      // 512 KB

  prep_wt   <<<(A_DIM * E_DIM) / 256, 256, 0, stream>>>(We, Wt);
  prep_wdb  <<<(B_DIM * A_DIM) / 256, 256, 0, stream>>>(dec, Wd, bd, be, wdb);
  score_kernel<<<S_DIM / 2, 128, LDS_BYTES, stream>>>(enc, Wt, wdb, Ws, scores);
  softmax_kernel<<<B_DIM, 256, 0, stream>>>(scores);
  ctx_partial<<<dim3(E_DIM / 256, B_DIM, 8), 256, 0, stream>>>(enc, scores, part);
  ctx_reduce <<<(B_DIM * E_DIM) / 256, 256, 0, stream>>>(part, out);
}
